// TeixidoInhibitoryLayer_6073083756906
// MI455X (gfx1250) — matrix operand order repaired and checked by reading, not compiled, not run on hardware
//
#include <hip/hip_runtime.h>
#include <math.h>

// ---------------------------------------------------------------------------
// out[b,o] = T*logsumexp_i( (gated_x[b,i] + w[o,i]) / T ), masked -> -1e9
//  == T*log( sum_i exp(g/T) * mask*exp(w/T) )   (exact factorization; masked
//     terms underflow to exactly 0 in the reference too)
// => f16 WMMA GEMM  S = Aexp @ Bexp^T,  out = T*log(S)
// ---------------------------------------------------------------------------

#define BATCH    64
#define N_IN     2048
#define N_OUT    2048
#define NSPLIT   8
#define KSLICE   (N_IN / NSPLIT)      // 256
#define TEMP_F   0.1f
#define INV_TEMP 10.0f
#define NEG_FILL_F (-1e9f)

typedef _Float16 v16h __attribute__((ext_vector_type(16)));
typedef _Float16 v8h  __attribute__((ext_vector_type(8)));
typedef float    v8f  __attribute__((ext_vector_type(8)));
typedef float    v4f  __attribute__((ext_vector_type(4)));

// ---------------- Kernel 1: global max(|x|) --------------------------------
__global__ __launch_bounds__(1024)
void k_absmax(const float* __restrict__ x, float* __restrict__ amax_out) {
    __shared__ float red[1024];
    float m = 0.0f;
    for (int i = threadIdx.x; i < BATCH * N_IN; i += 1024)
        m = fmaxf(m, fabsf(x[i]));
    red[threadIdx.x] = m;
    __syncthreads();
    for (int s = 512; s > 0; s >>= 1) {
        if (threadIdx.x < s)
            red[threadIdx.x] = fmaxf(red[threadIdx.x], red[threadIdx.x + s]);
        __syncthreads();
    }
    if (threadIdx.x == 0) amax_out[0] = red[0];
}

// ---------------- Kernel 2: consensus gate + A = f16(exp(g/T)) -------------
__global__ __launch_bounds__(256)
void k_prep_a(const float* __restrict__ x, const float* __restrict__ amax_in,
              _Float16* __restrict__ aexp) {
    __shared__ float red[256];
    __shared__ float cons_sh;
    const int b = blockIdx.x;
    const float inv = 1.0f / (amax_in[0] + 1e-6f);
    const float* xr = x + b * N_IN;
    const int per = N_IN / 256;               // 8
    const int base = threadIdx.x * per;

    float s = 0.0f;
    for (int j = 0; j < per; ++j) s += xr[base + j] * inv;
    red[threadIdx.x] = s;
    __syncthreads();
    for (int st = 128; st > 0; st >>= 1) {
        if (threadIdx.x < st) red[threadIdx.x] += red[threadIdx.x + st];
        __syncthreads();
    }
    if (threadIdx.x == 0) cons_sh = red[0] * (1.0f / (float)N_IN);
    __syncthreads();
    const float cons = cons_sh;

    for (int j = 0; j < per; ++j) {
        float xn = xr[base + j] * inv;
        float g  = (fabsf(xn - cons) < 1.0f) ? xn : 0.0f;   // EPSILON = 1.0
        aexp[b * N_IN + base + j] = (_Float16)__expf(g * INV_TEMP);
    }
}

// ---------------- Kernel 3: fused WMMA GEMM (split-K) ----------------------
// wave owns one N-tile (16 outputs) and all 4 M-tiles (M=64); B fragment
// (mask*exp(w/T)) computed on the fly and reused across the 4 WMMAs.
__global__ __launch_bounds__(256)
void k_gemm(const float* __restrict__ w, const float* __restrict__ mk,
            const _Float16* __restrict__ aexp, float* __restrict__ part) {
    const int lane  = threadIdx.x & 31;
    const int wave  = threadIdx.x >> 5;
    const int n0    = (blockIdx.x * 8 + wave) * 16;
    const int split = blockIdx.y;
    const int k0    = split * KSLICE;
    const int hi    = (lane >= 16);
    const int n     = n0 + (lane & 15);
    const int arow  = lane & 15;

    // Contiguous k order used here for A and B (wave32):
    //  A 16x32 f16: lane%16 = M, lane<16 -> K=0..15, lane>=16 -> K=16..31
    //  B 32x16 f16: lane%16 = N, lane<16 -> K=0..15, lane>=16 -> K=16..31
    const int ac0 = hi ? 16 : 0;
    const int ac1 = hi ? 24 : 8;
    const int bc  = hi ? 16 : 0;

    const float* wrow = w  + (size_t)n * N_IN;
    const float* mrow = mk + (size_t)n * N_IN;

    v8f acc[4] = {};

    for (int k = k0; k < k0 + KSLICE; k += 32) {
        // ---- B fragment: mask * exp(w/T), 16 consecutive K per lane -------
        const float* wp = wrow + k + bc;
        const float* mp = mrow + k + bc;
        __builtin_prefetch(wp + 32, 0, 0);     // global_prefetch_b8
        __builtin_prefetch(mp + 32, 0, 0);
        v16h bf;
        #pragma unroll
        for (int q = 0; q < 4; ++q) {
            v4f wq = ((const v4f*)wp)[q];
            v4f mq = ((const v4f*)mp)[q];
            #pragma unroll
            for (int j = 0; j < 4; ++j) {
                float val = (mq[j] != 0.0f) ? __expf(wq[j] * INV_TEMP) : 0.0f;
                bf[q * 4 + j] = (_Float16)val;
            }
        }
        // ---- 4 M-tiles: load A fragment, WMMA -----------------------------
        #pragma unroll
        for (int t = 0; t < 4; ++t) {
            const _Float16* ap = aexp + (size_t)(t * 16 + arow) * N_IN;
            v8h alo = *(const v8h*)(ap + k + ac0);
            v8h ahi = *(const v8h*)(ap + k + ac1);
            v16h af;
            #pragma unroll
            for (int j = 0; j < 8; ++j) { af[j] = alo[j]; af[8 + j] = ahi[j]; }
            acc[t] = __builtin_amdgcn_wmma_f32_16x16x32_f16(
                         false, af, false, bf, (short)0, acc[t], false, false);
        }
    }

    // ---- store per-split partials (C layout: vgpr r -> M=r / 8+r) ---------
    float* pbase = part + (size_t)split * BATCH * N_OUT;
    #pragma unroll
    for (int t = 0; t < 4; ++t) {
        #pragma unroll
        for (int r = 0; r < 8; ++r) {
            int m = t * 16 + (hi ? 8 + r : r);
            pbase[(size_t)m * N_OUT + n] = acc[t][r];
        }
    }
}

// ---------------- Kernel 4: fixed-order split reduce + T*log ---------------
__global__ __launch_bounds__(256)
void k_finish(const float* __restrict__ part, float* __restrict__ out) {
    int idx = blockIdx.x * 256 + threadIdx.x;
    if (idx >= BATCH * N_OUT) return;
    float s = 0.0f;
    #pragma unroll
    for (int sp = 0; sp < NSPLIT; ++sp)
        s += part[(size_t)sp * BATCH * N_OUT + idx];
    // all-masked row in the reference collapses to ~NEG_FILL
    out[idx] = (s > 1e-30f) ? (TEMP_F * logf(s)) : NEG_FILL_F;
}

// ---------------------------------------------------------------------------
extern "C" void kernel_launch(void* const* d_in, const int* in_sizes, int n_in,
                              void* d_out, int out_size, void* d_ws, size_t ws_size,
                              hipStream_t stream) {
    const float* x  = (const float*)d_in[0];
    const float* w  = (const float*)d_in[1];
    const float* mk = (const float*)d_in[2];
    float* out = (float*)d_out;

    char* ws = (char*)d_ws;
    float*     amax = (float*)(ws + 0);
    _Float16*  aexp = (_Float16*)(ws + 256);
    float*     part = (float*)(ws + 256 + BATCH * N_IN * 2);   // 256-aligned

    k_absmax<<<1, 1024, 0, stream>>>(x, amax);
    k_prep_a<<<BATCH, 256, 0, stream>>>(x, amax, aexp);
    k_gemm<<<dim3(N_OUT / (16 * 8), NSPLIT), 256, 0, stream>>>(w, mk, aexp, part);
    k_finish<<<(BATCH * N_OUT + 255) / 256, 256, 0, stream>>>(part, out);
}
